// GRUD_34333968564822
// MI455X (gfx1250) — matrix operand order repaired and checked by reading, not compiled, not run on hardware
//
#include <hip/hip_runtime.h>

typedef __attribute__((ext_vector_type(16))) __bf16 v16bf;
typedef __attribute__((ext_vector_type(8)))  __bf16 v8bf;
typedef __attribute__((ext_vector_type(8)))  float  v8f;

#define BB 512
#define TT 256
#define II 128
#define HH 512

// ---------------- workspace layout ----------------
// f32 region (offsets in floats)
constexpr size_t H0_OFF   = 0;                          // [512,512] h0 state
constexpr size_t H1_OFF   = H0_OFF + (size_t)BB * HH;   // [512,512] h1 state
constexpr size_t BIAS_OFF = H1_OFF + (size_t)BB * HH;
constexpr size_t BG_OFF   = BIAS_OFF;                   // 1152 (bdh0|bdh1|bdx0)
constexpr size_t B0R_OFF  = BIAS_OFF + 1152;
constexpr size_t B0U_OFF  = B0R_OFF + 512;
constexpr size_t B0IN_OFF = B0U_OFF + 512;
constexpr size_t B0HN_OFF = B0IN_OFF + 512;
constexpr size_t B1R_OFF  = B0HN_OFF + 512;
constexpr size_t B1U_OFF  = B1R_OFF + 512;
constexpr size_t B1IN_OFF = B1U_OFF + 512;
constexpr size_t B1HN_OFF = B1IN_OFF + 512;
constexpr size_t F32_TOTAL = B1HN_OFF + 512;            // 529,536 floats

// bf16 region (offsets in halves), starts at wsf + F32_TOTAL
constexpr size_t A0_OFF   = 0;                           // [512,640]  = [x_in | h0d]
constexpr size_t A1_OFF   = A0_OFF + (size_t)BB * 640;   // [512,1024] = [h0_new | h1d]
constexpr size_t PG_OFF   = A1_OFF + (size_t)BB * 1024;  // gamma weights [1152,128] packed
constexpr size_t P0R_OFF  = PG_OFF  + (size_t)1152 * 128;
constexpr size_t P0U_OFF  = P0R_OFF + (size_t)512 * 640;
constexpr size_t P0IN_OFF = P0U_OFF + (size_t)512 * 640;
constexpr size_t P0HN_OFF = P0IN_OFF + (size_t)512 * 128;
constexpr size_t P1R_OFF  = P0HN_OFF + (size_t)512 * 512;
constexpr size_t P1U_OFF  = P1R_OFF + (size_t)512 * 1024;
constexpr size_t P1IN_OFF = P1U_OFF + (size_t)512 * 1024;
constexpr size_t P1HN_OFF = P1IN_OFF + (size_t)512 * 512;

// ---------------- helpers ----------------
__device__ __forceinline__ v16bf a_frag_lds(const __bf16* lA, int K, int kb, int lane) {
  // A 16x32 bf16 layout: lane&15 -> row; lane<16 holds K {0..7,16..23}, lane>=16 holds {8..15,24..31}
  const __bf16* p = lA + (size_t)(lane & 15) * K + kb * 32 + ((lane >> 4) << 3);
  v8bf lo = *(const v8bf*)p;
  v8bf hi = *(const v8bf*)(p + 16);
  v16bf f;
#pragma unroll
  for (int i = 0; i < 8; ++i) { f[i] = lo[i]; f[i + 8] = hi[i]; }
  return f;
}

__device__ __forceinline__ v16bf b_frag(const __bf16* P, int KB, int nb, int kb, int lane) {
  // fragment-packed weights: contiguous 16 halves per lane per (nb,kb) block
  return *(const v16bf*)(P + (((size_t)nb * KB + kb) * 32 + lane) * (size_t)16);
}

__device__ __forceinline__ v8f wmma_bf16(v16bf a, v16bf b, v8f c) {
  return __builtin_amdgcn_wmma_f32_16x16x32_bf16(false, a, false, b, (short)0, c, false, false);
}

__device__ __forceinline__ float sigm(float x) { return 1.f / (1.f + __expf(-x)); }

// Async copy of `per_thread` 16-byte chunks per thread (global -> LDS), cooperatively issued by
// the calling thread at stride 256 chunks.  ASYNCcnt-tracked (CDNA5 path:
// GLOBAL_LOAD_ASYNC_TO_LDS_B128 bypasses VGPRs entirely).
__device__ __forceinline__ void async_stage_b128(const void* gsrc, void* lds_dst,
                                                 int tid, int per_thread) {
  unsigned ldsbase = (unsigned)(uintptr_t)lds_dst;  // LDS aperture: low 32 bits = LDS addr
#pragma unroll
  for (int i = 0; i < 8; ++i) {
    if (i >= per_thread) break;
    unsigned dstb = ldsbase + (unsigned)(tid + 256 * i) * 16u;
    const uint4* s = (const uint4*)gsrc + tid + 256 * i;
    asm volatile("global_load_async_to_lds_b128 %0, %1, off"
                 :: "v"(dstb), "v"(s) : "memory");
  }
  asm volatile("s_wait_asynccnt 0x0" ::: "memory");
}

// ---------------- one-time kernels ----------------
__global__ void zero_state_k(float* wsf) {
  size_t i = (size_t)blockIdx.x * blockDim.x + threadIdx.x;
  if (i < (size_t)2 * BB * HH) wsf[i] = 0.f;  // h0 then h1, contiguous
}

// Pack W rows [rowOff..rowOff+N) of (srcA cols 0..Ka)++(srcB cols 0..Kb) into
// WMMA B-fragment layout (K x N per ISA 7.12.2).
__global__ void pack_w_k(const float* __restrict__ srcA, int ldA, int rowOffA, int Ka,
                         const float* __restrict__ srcB, int ldB, int rowOffB, int Kb,
                         __bf16* __restrict__ dst, int N) {
  int K = Ka + Kb;
  int tid = blockIdx.x * blockDim.x + threadIdx.x;
  if (tid >= N * K) return;
  int n = tid / K, k = tid - n * K;
  float v = (k < Ka) ? srcA[(size_t)(rowOffA + n) * ldA + k]
                     : srcB[(size_t)(rowOffB + n) * ldB + (k - Ka)];
  int nb = n >> 4, kb = k >> 5, kk = k & 31;
  int lane = (((kk >> 3) & 1) ? 16 : 0) + (n & 15);
  int j = (kk & 7) + ((kk >> 4) << 3);
  dst[(((size_t)nb * (K >> 5) + kb) * 32 + lane) * 16 + j] = (__bf16)v;
}

__global__ void prep_bias_k(const float* bdh0, const float* bdh1, const float* bdx0,
                            const float* bih0, const float* bhh0,
                            const float* bih1, const float* bhh1, float* wsf) {
  int i = blockIdx.x * blockDim.x + threadIdx.x;
  if (i < 1152)
    wsf[BG_OFF + i] = (i < 512) ? bdh0[i] : (i < 1024 ? bdh1[i - 512] : bdx0[i - 1024]);
  if (i < 512) {
    wsf[B0R_OFF + i]  = bih0[i] + bhh0[i];
    wsf[B0U_OFF + i]  = bih0[512 + i] + bhh0[512 + i];
    wsf[B0IN_OFF + i] = bih0[1024 + i];
    wsf[B0HN_OFF + i] = bhh0[1024 + i];
    wsf[B1R_OFF + i]  = bih1[i] + bhh1[i];
    wsf[B1U_OFF + i]  = bih1[512 + i] + bhh1[512 + i];
    wsf[B1IN_OFF + i] = bih1[1024 + i];
    wsf[B1HN_OFF + i] = bhh1[1024 + i];
  }
}

// ---------------- per-step kernel S1: gammas + decayed states + x_in ----------------
// GEMM M=512, K=128, N=1152: cols [0,512)=delta@Wdh0 -> h0d, [512,1024)=delta@Wdh1 -> h1d,
// [1024,1152)=x@Wdx0 -> x_in.  grid(32 m-tiles, 9 n-groups of 128), 256 thr = 8 waves.
__global__ __launch_bounds__(256)
void step1_k(int t, const float* __restrict__ x, const float* __restrict__ mask,
             const float* __restrict__ delta, float* __restrict__ wsf,
             __bf16* __restrict__ wsh) {
  __shared__ __align__(16) __bf16 lA[16 * II];
  const int tid = threadIdx.x;
  const int m0 = blockIdx.x * 16;
  const int ngrp = blockIdx.y;
  const float* Asrc = (ngrp == 8) ? x : delta;
  {  // stage 16x128 f32 -> bf16 LDS (conversion in flight, so VALU path)
    int e = tid * 8, row = e >> 7, col = e & 127;
    const float4* p = (const float4*)(Asrc + (size_t)(m0 + row) * TT * II + (size_t)t * II + col);
    float4 u0 = p[0], u1 = p[1];
    __bf16* d = &lA[row * II + col];
    d[0] = (__bf16)u0.x; d[1] = (__bf16)u0.y; d[2] = (__bf16)u0.z; d[3] = (__bf16)u0.w;
    d[4] = (__bf16)u1.x; d[5] = (__bf16)u1.y; d[6] = (__bf16)u1.z; d[7] = (__bf16)u1.w;
  }
  __syncthreads();
  const int wave = tid >> 5, lane = tid & 31;
  const int n0 = ngrp * 128 + wave * 16;
  const int nb = n0 >> 4;
  const __bf16* Pg = wsh + PG_OFF;
  v8f acc = {0.f, 0.f, 0.f, 0.f, 0.f, 0.f, 0.f, 0.f};
#pragma unroll
  for (int kb = 0; kb < 4; ++kb)
    acc = wmma_bf16(a_frag_lds(lA, II, kb, lane), b_frag(Pg, 4, nb, kb, lane), acc);

  const int col = n0 + (lane & 15);
  const int mbase = m0 + ((lane >> 4) << 3);
  const float bg = wsf[BG_OFF + col];
  float* h0s = wsf + H0_OFF;
  float* h1s = wsf + H1_OFF;
  __bf16* A0 = wsh + A0_OFF;
  __bf16* A1 = wsh + A1_OFF;
#pragma unroll
  for (int r = 0; r < 8; ++r) {
    int m = mbase + r;
    float g = __expf(-fmaxf(acc[r] + bg, 0.f));
    if (col < 512) {
      A0[(size_t)m * 640 + 128 + col] = (__bf16)(g * h0s[(size_t)m * HH + col]);
    } else if (col < 1024) {
      int c = col - 512;
      A1[(size_t)m * 1024 + 512 + c] = (__bf16)(g * h1s[(size_t)m * HH + c]);
    } else {
      int c = col - 1024;
      size_t idx = (size_t)m * TT * II + (size_t)t * II + c;
      float xv = x[idx], mv = mask[idx];
      A0[(size_t)m * 640 + c] = (__bf16)(mv * xv + (1.f - mv) * g * xv);
    }
  }
}

// ---------------- per-step kernel S2: layer-0 fused gate GEMM -> h0_new ----------------
// A = [x_in | h0d] (K=640). r,u over full K; i_n over K[0,128); h_n over K[128,640).
__global__ __launch_bounds__(256)
void step2_k(float* __restrict__ wsf, __bf16* __restrict__ wsh) {
  __shared__ __align__(16) __bf16 lA[16 * 640];
  const int tid = threadIdx.x;
  const int m0 = blockIdx.x * 16;
  const int n0 = blockIdx.y * 128 + (tid >> 5) * 16;
  const __bf16* A0 = wsh + A0_OFF;
  // async-stage contiguous 16x640 bf16 A rows (1280 x b128, 5 per thread)
  async_stage_b128(A0 + (size_t)m0 * 640, lA, tid, 5);
  __syncthreads();
  const int lane = tid & 31, nb = n0 >> 4;
  const __bf16* PR = wsh + P0R_OFF;
  const __bf16* PU = wsh + P0U_OFF;
  const __bf16* PI = wsh + P0IN_OFF;
  const __bf16* PH = wsh + P0HN_OFF;
  v8f aR = {0.f,0.f,0.f,0.f,0.f,0.f,0.f,0.f}, aU = aR, aI = aR, aH = aR;
#pragma unroll 4
  for (int kb = 0; kb < 20; ++kb) {
    v16bf a = a_frag_lds(lA, 640, kb, lane);
    if (kb < 19)
      __builtin_prefetch((const void*)(PR + (((size_t)nb * 20 + kb + 1) * 32 + lane) * 16), 0, 3);
    aR = wmma_bf16(a, b_frag(PR, 20, nb, kb, lane), aR);
    aU = wmma_bf16(a, b_frag(PU, 20, nb, kb, lane), aU);
    if (kb < 4) aI = wmma_bf16(a, b_frag(PI, 4, nb, kb, lane), aI);
    else        aH = wmma_bf16(a, b_frag(PH, 16, nb, kb - 4, lane), aH);
  }
  const int n = n0 + (lane & 15);
  const int mbase = m0 + ((lane >> 4) << 3);
  const float br = wsf[B0R_OFF + n], bu = wsf[B0U_OFF + n];
  const float bi = wsf[B0IN_OFF + n], bh = wsf[B0HN_OFF + n];
  float* h0s = wsf + H0_OFF;
  __bf16* A1 = wsh + A1_OFF;
#pragma unroll
  for (int r = 0; r < 8; ++r) {
    int m = mbase + r;
    float hd = (float)A0[(size_t)m * 640 + 128 + n];     // h0d
    float rr = sigm(aR[r] + br);
    float uu = sigm(aU[r] + bu);
    float nn = tanhf(aI[r] + bi + rr * (aH[r] + bh));
    float hnew = (1.f - uu) * nn + uu * hd;
    h0s[(size_t)m * HH + n] = hnew;                      // state for next step
    A1[(size_t)m * 1024 + n] = (__bf16)hnew;             // feeds layer 1
  }
}

// ---------------- per-step kernel S3: layer-1 fused gate GEMM -> h1_new ----------------
// A = [h0_new | h1d] (K=1024). i_n over K[0,512); h_n over K[512,1024).
__global__ __launch_bounds__(256)
void step3_k(float* __restrict__ wsf, __bf16* __restrict__ wsh) {
  __shared__ __align__(16) __bf16 lA[16 * 1024];
  const int tid = threadIdx.x;
  const int m0 = blockIdx.x * 16;
  const int n0 = blockIdx.y * 128 + (tid >> 5) * 16;
  const __bf16* A1 = wsh + A1_OFF;
  // async-stage contiguous 16x1024 bf16 A rows (2048 x b128, 8 per thread)
  async_stage_b128(A1 + (size_t)m0 * 1024, lA, tid, 8);
  __syncthreads();
  const int lane = tid & 31, nb = n0 >> 4;
  const __bf16* PR = wsh + P1R_OFF;
  const __bf16* PU = wsh + P1U_OFF;
  const __bf16* PI = wsh + P1IN_OFF;
  const __bf16* PH = wsh + P1HN_OFF;
  v8f aR = {0.f,0.f,0.f,0.f,0.f,0.f,0.f,0.f}, aU = aR, aI = aR, aH = aR;
#pragma unroll 4
  for (int kb = 0; kb < 32; ++kb) {
    v16bf a = a_frag_lds(lA, 1024, kb, lane);
    if (kb < 31)
      __builtin_prefetch((const void*)(PR + (((size_t)nb * 32 + kb + 1) * 32 + lane) * 16), 0, 3);
    aR = wmma_bf16(a, b_frag(PR, 32, nb, kb, lane), aR);
    aU = wmma_bf16(a, b_frag(PU, 32, nb, kb, lane), aU);
    if (kb < 16) aI = wmma_bf16(a, b_frag(PI, 16, nb, kb, lane), aI);
    else         aH = wmma_bf16(a, b_frag(PH, 16, nb, kb - 16, lane), aH);
  }
  const int n = n0 + (lane & 15);
  const int mbase = m0 + ((lane >> 4) << 3);
  const float br = wsf[B1R_OFF + n], bu = wsf[B1U_OFF + n];
  const float bi = wsf[B1IN_OFF + n], bh = wsf[B1HN_OFF + n];
  float* h1s = wsf + H1_OFF;
#pragma unroll
  for (int r = 0; r < 8; ++r) {
    int m = mbase + r;
    float hd = (float)A1[(size_t)m * 1024 + 512 + n];    // h1d
    float rr = sigm(aR[r] + br);
    float uu = sigm(aU[r] + bu);
    float nn = tanhf(aI[r] + bi + rr * (aH[r] + bh));
    h1s[(size_t)m * HH + n] = (1.f - uu) * nn + uu * hd;
  }
}

// ---------------- epilogue kernels ----------------
__global__ __launch_bounds__(256)
void fc_k(const float* __restrict__ wsf, const float* __restrict__ wfc,
          const float* __restrict__ bfc, float* __restrict__ out) {
  int wave = threadIdx.x >> 5, lane = threadIdx.x & 31;
  int row = blockIdx.x * 8 + wave;
  const float* h1s = wsf + H1_OFF;
  float s = 0.f;
#pragma unroll
  for (int j = 0; j < 16; ++j)
    s += h1s[(size_t)row * HH + j * 32 + lane] * wfc[j * 32 + lane];
#pragma unroll
  for (int off = 16; off > 0; off >>= 1) s += __shfl_xor(s, off, 32);
  if (lane == 0) out[row] = s + bfc[0];
}

__global__ void copy_h_k(const float* __restrict__ wsf, float* __restrict__ out) {
  size_t i = (size_t)blockIdx.x * blockDim.x + threadIdx.x;
  if (i < (size_t)BB * HH) {
    out[512 + i] = wsf[H0_OFF + i];
    out[512 + (size_t)BB * HH + i] = wsf[H1_OFF + i];
  }
}

// ---------------- host ----------------
extern "C" void kernel_launch(void* const* d_in, const int* in_sizes, int n_in,
                              void* d_out, int out_size, void* d_ws, size_t ws_size,
                              hipStream_t stream) {
  const float* x     = (const float*)d_in[0];
  const float* mask  = (const float*)d_in[1];
  const float* delta = (const float*)d_in[2];
  const float* W_ih0 = (const float*)d_in[3];
  const float* b_ih0 = (const float*)d_in[4];
  const float* W_hh0 = (const float*)d_in[5];
  const float* b_hh0 = (const float*)d_in[6];
  const float* Wdh0  = (const float*)d_in[7];
  const float* bdh0  = (const float*)d_in[8];
  const float* Wdx0  = (const float*)d_in[9];
  const float* bdx0  = (const float*)d_in[10];
  const float* W_ih1 = (const float*)d_in[11];
  const float* b_ih1 = (const float*)d_in[12];
  const float* W_hh1 = (const float*)d_in[13];
  const float* b_hh1 = (const float*)d_in[14];
  const float* Wdh1  = (const float*)d_in[15];
  const float* bdh1  = (const float*)d_in[16];
  const float* Wfc   = (const float*)d_in[17];
  const float* bfc   = (const float*)d_in[18];
  float* out = (float*)d_out;
  float* wsf = (float*)d_ws;
  __bf16* wsh = (__bf16*)(wsf + F32_TOTAL);

  hipLaunchKernelGGL(zero_state_k, dim3(2048), dim3(256), 0, stream, wsf);

  auto packN = [&](const float* A, int ldA, int roA, int Ka,
                   const float* Bm, int ldB, int roB, int Kb, size_t dstOff, int N) {
    int total = N * (Ka + Kb);
    hipLaunchKernelGGL(pack_w_k, dim3((total + 255) / 256), dim3(256), 0, stream,
                       A, ldA, roA, Ka, Bm, ldB, roB, Kb, wsh + dstOff, N);
  };
  // gamma weights (N-offsets: per nb stride = (K/32)*512 = 2048 halves)
  packN(Wdh0, 128, 0, 128, Wdh0, 128, 0, 0, PG_OFF, 512);
  packN(Wdh1, 128, 0, 128, Wdh1, 128, 0, 0, PG_OFF + (size_t)32 * 2048, 512);
  packN(Wdx0, 128, 0, 128, Wdx0, 128, 0, 0, PG_OFF + (size_t)64 * 2048, 128);
  // layer 0 gates
  packN(W_ih0, 128, 0, 128,    W_hh0, 512, 0, 512,   P0R_OFF, 512);
  packN(W_ih0, 128, 512, 128,  W_hh0, 512, 512, 512, P0U_OFF, 512);
  packN(W_ih0, 128, 1024, 128, W_ih0, 128, 0, 0,     P0IN_OFF, 512);
  packN(W_hh0, 512, 1024, 512, W_hh0, 512, 0, 0,     P0HN_OFF, 512);
  // layer 1 gates
  packN(W_ih1, 512, 0, 512,    W_hh1, 512, 0, 512,   P1R_OFF, 512);
  packN(W_ih1, 512, 512, 512,  W_hh1, 512, 512, 512, P1U_OFF, 512);
  packN(W_ih1, 512, 1024, 512, W_ih1, 512, 0, 0,     P1IN_OFF, 512);
  packN(W_hh1, 512, 1024, 512, W_hh1, 512, 0, 0,     P1HN_OFF, 512);

  hipLaunchKernelGGL(prep_bias_k, dim3(5), dim3(256), 0, stream,
                     bdh0, bdh1, bdx0, b_ih0, b_hh0, b_ih1, b_hh1, wsf);

  for (int t = 0; t < TT; ++t) {
    hipLaunchKernelGGL(step1_k, dim3(32, 9), dim3(256), 0, stream, t, x, mask, delta, wsf, wsh);
    hipLaunchKernelGGL(step2_k, dim3(32, 4), dim3(256), 0, stream, wsf, wsh);
    hipLaunchKernelGGL(step3_k, dim3(32, 4), dim3(256), 0, stream, wsf, wsh);
  }

  hipLaunchKernelGGL(fc_k, dim3(64), dim3(256), 0, stream, wsf, Wfc, bfc, out);
  hipLaunchKernelGGL(copy_h_k, dim3(1024), dim3(256), 0, stream, wsf, out);
}
